// GraphformerLayer_15479062135313
// MI455X (gfx1250) — compile-verified, matrix operand order checked by reading, not run on hardware
//
#include <hip/hip_runtime.h>
#include <hip/hip_bf16.h>

typedef __attribute__((ext_vector_type(8)))  float  v8f;
typedef __attribute__((ext_vector_type(8)))  __bf16 v8bf;
typedef __attribute__((ext_vector_type(16))) __bf16 v16bf;

union Frag16 { v16bf v; v8bf h[2]; };

static __device__ __forceinline__ v8f wmma_bf16(v16bf a, v16bf b, v8f c) {
  return __builtin_amdgcn_wmma_f32_16x16x32_bf16(false, a, false, b, (short)0, c,
                                                 false, false);
}

// Load a 16x32 (rows x k) bf16 A/B fragment in the CDNA5 WMMA register layout:
// lanes 0-15 -> row r0+lane,    K = k0+{0..7, 16..23}
// lanes16-31 -> row r0+lane-16, K = k0+{8..15,24..31}
static __device__ __forceinline__ v16bf load_frag(const __bf16* mat, int ld, int r0,
                                                  int k0, int lane) {
  const int row = r0 + (lane & 15);
  const int kb  = k0 + ((lane >> 4) << 3);
  const __bf16* p = mat + (long)row * ld + kb;
  Frag16 f;
  f.h[0] = *reinterpret_cast<const v8bf*>(p);
  f.h[1] = *reinterpret_cast<const v8bf*>(p + 16);
  return f.v;
}

// CDNA5 async global->LDS copy of 16 bytes (tracked by ASYNCcnt).
static __device__ __forceinline__ void async_copy_b128(const __bf16* gptr,
                                                       __bf16* lptr) {
  const unsigned lds_off = (unsigned)(uintptr_t)lptr;  // addr[31:0] == LDS offset
  asm volatile("global_load_async_to_lds_b128 %0, %1, off"
               :: "v"(lds_off), "v"(gptr) : "memory");
}
#define ASYNC_WAIT_4() asm volatile("s_wait_asynccnt 0x4" ::: "memory")
#define ASYNC_WAIT_0() asm volatile("s_wait_asynccnt 0x0" ::: "memory")

// ---------------------------------------------------------------- convert
__global__ void cvt_f32_bf16(const float* __restrict__ in, __bf16* __restrict__ out,
                             int n) {
  int i = blockIdx.x * blockDim.x + threadIdx.x;
  if (i < n) out[i] = (__bf16)in[i];
}

// ---------------------------------------------------------------- QKV GEMM
// C = x(4096x512) @ W^T(512x512); Q scaled by 1/sqrt(64); V stored transposed.
// Q,K: [head][n][64]   Vt: [head][64][n]   (bf16)
__global__ __launch_bounds__(128) void qkv_gemm(const __bf16* __restrict__ xb,
                                                const __bf16* __restrict__ wq,
                                                const __bf16* __restrict__ wk,
                                                const __bf16* __restrict__ wv,
                                                __bf16* __restrict__ Qh,
                                                __bf16* __restrict__ Kh,
                                                __bf16* __restrict__ Vt) {
  const int lane = threadIdx.x & 31;
  const int wave = threadIdx.x >> 5;
  const int m0 = (blockIdx.x * 4 + wave) * 16;   // token rows
  const int n0 = blockIdx.y * 64;                // output cols (hd)
  const int z  = blockIdx.z;                     // 0=Q 1=K 2=V
  const __bf16* W = (z == 0) ? wq : (z == 1) ? wk : wv;

  v8f acc[4] = {};
  for (int k0 = 0; k0 < 512; k0 += 32) {
    v16bf a = load_frag(xb, 512, m0, k0, lane);
#pragma unroll
    for (int nt = 0; nt < 4; ++nt) {
      v16bf b = load_frag(W, 512, n0 + nt * 16, k0, lane);
      acc[nt] = wmma_bf16(a, b, acc[nt]);
    }
  }

  // D layout: lanes0-15 N=lane, M=reg; lanes16-31 N=lane-16, M=8+reg
#pragma unroll
  for (int nt = 0; nt < 4; ++nt) {
    const int o = n0 + nt * 16 + (lane & 15);    // hd index
    const int h = o >> 6, d = o & 63;
#pragma unroll
    for (int r = 0; r < 8; ++r) {
      const int m = m0 + ((lane < 16) ? r : 8 + r);
      const float v = acc[nt][r];
      if (z == 0)      Qh[((long)(h << 12) + m) * 64 + d] = (__bf16)(v * 0.125f);
      else if (z == 1) Kh[((long)(h << 12) + m) * 64 + d] = (__bf16)v;
      else             Vt[(long)o * 4096 + m] = (__bf16)v;   // o == h*64+d
    }
  }
}

// ---------------------------------------------------------------- attention
// 4 waves/block, one head per block-row; each wave owns a 16-query tile.
// K/V tiles (32 keys) are shared by all 4 waves -> staged in LDS with
// double-buffered async global->LDS copies (ASYNCcnt).
// Computes S^T = K @ Q^T so softmax reduction is in-lane + one xor16 shuffle,
// and the two 16-key S^T tiles form a native K=32 B-fragment of P^T.
__global__ __launch_bounds__(128) void attn_flash(const __bf16* __restrict__ Qh,
                                                  const __bf16* __restrict__ Kh,
                                                  const __bf16* __restrict__ Vt,
                                                  __bf16* __restrict__ Ob) {
  __shared__ __bf16 Ks[2][32 * 64];   // [key][d]
  __shared__ __bf16 Vs[2][64 * 32];   // [d][key]

  const int tid  = threadIdx.x;
  const int lane = tid & 31;
  const int wave = tid >> 5;
  const int h  = blockIdx.y;
  const int q0 = (blockIdx.x * 4 + wave) * 16;
  const __bf16* Q = Qh + (long)(h << 12) * 64;
  const __bf16* K = Kh + (long)(h << 12) * 64;
  const __bf16* V = Vt + (long)(h << 6) * 4096;

  // loop-invariant Q B-fragments (16 queries x 64 d)
  const v16bf qb0 = load_frag(Q, 64, q0, 0, lane);
  const v16bf qb1 = load_frag(Q, 64, q0, 32, lane);

  // stage kt=0 into buffer 0: 4 async b128 per thread per stage
  {
    async_copy_b128(K + 0 * 64 + tid * 8,        Ks[0] + tid * 8);
    async_copy_b128(K + 0 * 64 + (tid + 128) * 8, Ks[0] + (tid + 128) * 8);
    const int c0 = tid, c1 = tid + 128;
    async_copy_b128(V + (c0 >> 2) * 4096 + 0 + (c0 & 3) * 8, Vs[0] + c0 * 8);
    async_copy_b128(V + (c1 >> 2) * 4096 + 0 + (c1 & 3) * 8, Vs[0] + c1 * 8);
  }

  v8f o0 = {}, o1 = {}, o2 = {}, o3 = {};
  float mrun = -3.0e38f, lrun = 0.0f;

  for (int it = 0; it < 128; ++it) {
    const int kt  = it * 32;
    const int cur = it & 1;

    if (it + 1 < 128) {  // prefetch next 32-key tile into the other buffer
      const int kn = kt + 32;
      __bf16* kd = Ks[cur ^ 1];
      __bf16* vd = Vs[cur ^ 1];
      async_copy_b128(K + kn * 64 + tid * 8,         kd + tid * 8);
      async_copy_b128(K + kn * 64 + (tid + 128) * 8, kd + (tid + 128) * 8);
      const int c0 = tid, c1 = tid + 128;
      async_copy_b128(V + (c0 >> 2) * 4096 + kn + (c0 & 3) * 8, vd + c0 * 8);
      async_copy_b128(V + (c1 >> 2) * 4096 + kn + (c1 & 3) * 8, vd + c1 * 8);
      ASYNC_WAIT_4();   // current buffer's 4 copies (issued last iter) are done
    } else {
      ASYNC_WAIT_0();
    }
    __syncthreads();    // all waves' copies into buf[cur] visible

    const __bf16* kc = Ks[cur];
    const __bf16* vc = Vs[cur];

    // S^T tiles from LDS: keys 0..15 and 16..31 of this tile
    v16bf ka0 = load_frag(kc, 64, 0,  0,  lane);
    v16bf ka1 = load_frag(kc, 64, 0,  32, lane);
    v16bf kb0 = load_frag(kc, 64, 16, 0,  lane);
    v16bf kb1 = load_frag(kc, 64, 16, 32, lane);
    v8f s0 = {}; s0 = wmma_bf16(ka0, qb0, s0); s0 = wmma_bf16(ka1, qb1, s0);
    v8f s1 = {}; s1 = wmma_bf16(kb0, qb0, s1); s1 = wmma_bf16(kb1, qb1, s1);

    // online softmax over the key axis (regs + lane-half exchange)
    float tmax = s0[0];
#pragma unroll
    for (int r = 0; r < 8; ++r) { tmax = fmaxf(tmax, s0[r]); tmax = fmaxf(tmax, s1[r]); }
    tmax = fmaxf(tmax, __shfl_xor(tmax, 16, 32));
    const float mnew  = fmaxf(mrun, tmax);
    const float alpha = __expf(mrun - mnew);

    Frag16 p;
    float psum = 0.0f;
#pragma unroll
    for (int r = 0; r < 8; ++r) {
      const float e0 = __expf(s0[r] - mnew);
      const float e1 = __expf(s1[r] - mnew);
      psum += e0 + e1;
      p.v[r]     = (__bf16)e0;   // keys kt+{0..7 | 8..15}  (per lane half)
      p.v[8 + r] = (__bf16)e1;   // keys kt+{16..23 | 24..31}
    }
    psum += __shfl_xor(psum, 16, 32);
    lrun = lrun * alpha + psum;
    mrun = mnew;
#pragma unroll
    for (int r = 0; r < 8; ++r) {
      o0[r] *= alpha; o1[r] *= alpha; o2[r] *= alpha; o3[r] *= alpha;
    }

    // O^T += V^T(64 x 32keys) @ P^T(32keys x 16 queries), V from LDS
    v16bf va0 = load_frag(vc, 32, 0,  0, lane);
    v16bf va1 = load_frag(vc, 32, 16, 0, lane);
    v16bf va2 = load_frag(vc, 32, 32, 0, lane);
    v16bf va3 = load_frag(vc, 32, 48, 0, lane);
    o0 = wmma_bf16(va0, p.v, o0);
    o1 = wmma_bf16(va1, p.v, o1);
    o2 = wmma_bf16(va2, p.v, o2);
    o3 = wmma_bf16(va3, p.v, o3);

    __syncthreads();    // everyone done reading buf[cur] before it's re-staged
  }

  const float inv = 1.0f / lrun;
  const int q = q0 + (lane & 15);
#pragma unroll
  for (int r = 0; r < 8; ++r) {
    const int dr = (lane < 16) ? r : 8 + r;
    Ob[(long)q * 512 + (h << 6) +  0 + dr] = (__bf16)(o0[r] * inv);
    Ob[(long)q * 512 + (h << 6) + 16 + dr] = (__bf16)(o1[r] * inv);
    Ob[(long)q * 512 + (h << 6) + 32 + dr] = (__bf16)(o2[r] * inv);
    Ob[(long)q * 512 + (h << 6) + 48 + dr] = (__bf16)(o3[r] * inv);
  }
}

// ---------------------------------------------------------------- output proj
// out(4096x64) = Ob(4096x512) @ Wo^T(512x64) + bo, f32 output.
__global__ __launch_bounds__(128) void out_gemm(const __bf16* __restrict__ Ob,
                                                const __bf16* __restrict__ Wob,
                                                const float* __restrict__ bo,
                                                float* __restrict__ out) {
  const int lane = threadIdx.x & 31;
  const int wave = threadIdx.x >> 5;
  const int m0 = (blockIdx.x * 4 + wave) * 16;
  const int o0 = blockIdx.y * 16;

  v8f acc = {};
  for (int k0 = 0; k0 < 512; k0 += 32) {
    v16bf a = load_frag(Ob, 512, m0, k0, lane);
    v16bf b = load_frag(Wob, 512, o0, k0, lane);
    acc = wmma_bf16(a, b, acc);
  }
  const int o = o0 + (lane & 15);
  const float bias = bo[o];
#pragma unroll
  for (int r = 0; r < 8; ++r) {
    const int m = m0 + ((lane < 16) ? r : 8 + r);
    out[(long)m * 64 + o] = acc[r] + bias;
  }
}

// ---------------------------------------------------------------- launch
extern "C" void kernel_launch(void* const* d_in, const int* in_sizes, int n_in,
                              void* d_out, int out_size, void* d_ws, size_t ws_size,
                              hipStream_t stream) {
  const float* x  = (const float*)d_in[0];   // 4096x512
  const float* Wq = (const float*)d_in[1];   // 512x512
  const float* Wk = (const float*)d_in[2];   // 512x512
  const float* Wv = (const float*)d_in[3];   // 512x512
  const float* Wo = (const float*)d_in[4];   // 64x512
  const float* bo = (const float*)d_in[5];   // 64
  float* out = (float*)d_out;                // 4096x64

  char* ws = (char*)d_ws;
  size_t off = 0;
  __bf16* xb  = (__bf16*)(ws + off); off += (size_t)4096 * 512 * 2;  // 4 MB
  __bf16* wqb = (__bf16*)(ws + off); off += (size_t)512 * 512 * 2;
  __bf16* wkb = (__bf16*)(ws + off); off += (size_t)512 * 512 * 2;
  __bf16* wvb = (__bf16*)(ws + off); off += (size_t)512 * 512 * 2;
  __bf16* wob = (__bf16*)(ws + off); off += (size_t)64 * 512 * 2;
  __bf16* Qh  = (__bf16*)(ws + off); off += (size_t)8 * 4096 * 64 * 2;
  __bf16* Kh  = (__bf16*)(ws + off); off += (size_t)8 * 4096 * 64 * 2;
  __bf16* Vt  = (__bf16*)(ws + off); off += (size_t)8 * 64 * 4096 * 2;
  __bf16* Ob  = (__bf16*)(ws + off); off += (size_t)4096 * 512 * 2;

  cvt_f32_bf16<<<(4096 * 512 + 255) / 256, 256, 0, stream>>>(x, xb, 4096 * 512);
  cvt_f32_bf16<<<(512 * 512 + 255) / 256, 256, 0, stream>>>(Wq, wqb, 512 * 512);
  cvt_f32_bf16<<<(512 * 512 + 255) / 256, 256, 0, stream>>>(Wk, wkb, 512 * 512);
  cvt_f32_bf16<<<(512 * 512 + 255) / 256, 256, 0, stream>>>(Wv, wvb, 512 * 512);
  cvt_f32_bf16<<<(64 * 512 + 255) / 256, 256, 0, stream>>>(Wo, wob, 64 * 512);

  qkv_gemm<<<dim3(64, 8, 3), 128, 0, stream>>>(xb, wqb, wkb, wvb, Qh, Kh, Vt);
  attn_flash<<<dim3(64, 8), 128, 0, stream>>>(Qh, Kh, Vt, Ob);
  out_gemm<<<dim3(64, 4), 128, 0, stream>>>(Ob, wob, bo, out);
}
